// CausalSelfAttention_masked_for_formula_274877907729
// MI455X (gfx1250) — compile-verified, matrix operand order checked by reading, not run on hardware
//
#include <hip/hip_runtime.h>
#include <hip/hip_bf16.h>
#include <math.h>

// ---------------- problem constants ----------------
#define Bn   8
#define Tn   512
#define Cn   768
#define Hn   12
#define HDn  64          // head dim
#define BT   (Bn*Tn)     // 4096
#define ADJF 12

typedef __attribute__((ext_vector_type(16))) _Float16 v16h;
typedef __attribute__((ext_vector_type(8)))  _Float16 v8h;
typedef __attribute__((ext_vector_type(4)))  _Float16 v4h;
typedef __attribute__((ext_vector_type(8)))  float    v8f;
typedef __attribute__((ext_vector_type(4)))  float    v4f;

union Frag { v16h v; v8h h2[2]; };

// Build an A/B WMMA f16 fragment from a row-major f16 matrix.
// ISA layout per lane (lane = hlf*16 + r16): row/col = r16,
// K chunks at [hlf*8 .. hlf*8+8) and [hlf*8+16 .. hlf*8+24).
__device__ inline v16h load_frag_h(const _Float16* base /* row start + k0 + hlf*8 */) {
    Frag u;
    u.h2[0] = *(const v8h*)(base);
    u.h2[1] = *(const v8h*)(base + 16);
    return u.v;
}

// CDNA5 async global->LDS copy (16B per lane), tracked by ASYNCcnt.
// LDS dest address = low 32 bits of the flat address (ISA 10.2 aperture rule).
__device__ inline void async_copy16(void* lds_dst, const void* gsrc) {
    const unsigned int lo = (unsigned int)(uintptr_t)lds_dst;
    asm volatile("global_load_async_to_lds_b128 %0, %1, off"
                 :: "v"(lo), "v"(gsrc) : "memory");
}
__device__ inline void async_wait0() {
    asm volatile("s_wait_asynccnt 0x0" ::: "memory");
}

// ---------------------------------------------------------------------------
// Kernel 0: elementwise f32 -> f16 (x and weight pre-conversion, read once)
// ---------------------------------------------------------------------------
__global__ __launch_bounds__(256) void cvt_f16_kernel(
    const float* __restrict__ in, _Float16* __restrict__ out, int n4)
{
    const int i = blockIdx.x * 256 + threadIdx.x;
    if (i >= n4) return;
    v4f f = *(const v4f*)(in + (size_t)i * 4);
    v4h h;
#pragma unroll
    for (int j = 0; j < 4; ++j) h[j] = (_Float16)f[j];
    *(v4h*)(out + (size_t)i * 4) = h;
}

// ---------------------------------------------------------------------------
// Kernel 1: adjacency bias  AF[b,h,q,k] = A2[h,:] . gelu(A1 @ adj[b,q,k,:])
// One thread per (b,q,k) pair; adj read is 48B contiguous per thread.
// ---------------------------------------------------------------------------
__global__ __launch_bounds__(256) void adjbias_kernel(
    const float* __restrict__ adj,      // [B,T,T,12]
    const float* __restrict__ A1,       // [24,12]
    const float* __restrict__ A2,       // [12,24]
    _Float16*    __restrict__ AF)       // [B,H,T,T] f16
{
    __shared__ float sA1[24 * 12];
    __shared__ float sA2[12 * 24];
    const int tid = threadIdx.x;
    for (int i = tid; i < 288; i += 256) sA1[i] = A1[i];
    for (int i = tid; i < 288; i += 256) sA2[i] = A2[i];
    __syncthreads();

    const size_t g = (size_t)blockIdx.x * 256 + tid;   // over B*T*T
    const float* av = adj + g * ADJF;
    float v[12];
    v4f t0 = *(const v4f*)(av);
    v4f t1 = *(const v4f*)(av + 4);
    v4f t2 = *(const v4f*)(av + 8);
#pragma unroll
    for (int i = 0; i < 4; ++i) { v[i] = t0[i]; v[4 + i] = t1[i]; v[8 + i] = t2[i]; }

    float gv[24];
#pragma unroll
    for (int e = 0; e < 24; ++e) {
        float s = 0.f;
#pragma unroll
        for (int f = 0; f < 12; ++f) s += sA1[e * 12 + f] * v[f];
        gv[e] = 0.5f * s * (1.0f + erff(s * 0.70710678118654752f)); // exact gelu
    }

    const size_t bq = g >> 9;          // b*T + q
    const size_t k  = g & 511;
    const size_t b  = bq >> 9;
    const size_t q  = bq & 511;
#pragma unroll
    for (int h = 0; h < Hn; ++h) {
        float o = 0.f;
#pragma unroll
        for (int e = 0; e < 24; ++e) o += sA2[h * 24 + e] * gv[e];
        AF[(((b * Hn + h) * Tn + q) * Tn) + k] = (_Float16)o;
    }
}

// ---------------------------------------------------------------------------
// Kernel 2: QKV projection.  O = Xh @ Wh^T + bias   (f16 in, f16 out)
// Block: 256 threads / 8 waves, 64x64 output tile, 2 WMMA D-tiles per wave.
// MODE 0: store [B,H,T,64] row-major (Q, K)
// MODE 1: store [B,H,64,T]  (V transposed, so attn B-fragments are contiguous)
// ---------------------------------------------------------------------------
template <int MODE>
__global__ __launch_bounds__(256) void gemm_qkv(
    const _Float16* __restrict__ X,     // [4096,768] f16
    const _Float16* __restrict__ W,     // [768,768] f16 (out,in)
    const float*    __restrict__ bias,  // [768]
    _Float16*       __restrict__ out)
{
    const int m0blk = blockIdx.x * 64;
    const int n0blk = blockIdx.y * 64;
    const int tid = threadIdx.x;
    const int wave = tid >> 5, lane = tid & 31;
    const int r16 = lane & 15, hlf = lane >> 4;
    const int mt  = wave & 3;          // M-tile 0..3
    const int ntb = (wave >> 2) * 2;   // N-tiles {0,1} or {2,3}

    const _Float16* arow  = X + (size_t)(m0blk + mt * 16 + r16) * Cn + hlf * 8;
    const _Float16* brow0 = W + (size_t)(n0blk + ntb * 16 + r16) * Cn + hlf * 8;
    const _Float16* brow1 = brow0 + (size_t)16 * Cn;

    v8f acc[2] = {v8f{}, v8f{}};
#pragma unroll 4
    for (int kt = 0; kt < Cn / 32; ++kt) {
        const int k0 = kt * 32;
        v16h a  = load_frag_h(arow + k0);
        v16h b0 = load_frag_h(brow0 + k0);
        v16h b1 = load_frag_h(brow1 + k0);
        acc[0] = __builtin_amdgcn_wmma_f32_16x16x32_f16(
                     false, a, false, b0, (short)0, acc[0], false, false);
        acc[1] = __builtin_amdgcn_wmma_f32_16x16x32_f16(
                     false, a, false, b1, (short)0, acc[1], false, false);
    }
#pragma unroll
    for (int i = 0; i < 2; ++i) {
        const int col = n0blk + (ntb + i) * 16 + r16;
        const float bv = bias[col];
        const int hh = col >> 6, d = col & 63;
#pragma unroll
        for (int r = 0; r < 8; ++r) {
            const int rowg = m0blk + mt * 16 + hlf * 8 + r;   // 0..4095
            const int bb = rowg >> 9, t = rowg & 511;
            const float val = acc[i][r] + bv;
            size_t idx;
            if (MODE == 0) idx = (((size_t)bb * Hn + hh) * Tn + t) * HDn + d;
            else           idx = (((size_t)bb * Hn + hh) * HDn + d) * Tn + t;
            out[idx] = (_Float16)val;
        }
    }
}

// ---------------------------------------------------------------------------
// Kernel 3: fused attention per (b,h, 64-query tile).
//   S = Q K^T (WMMA, f16 into LDS) -> masked f32 softmax + 50*AF -> P (f16 LDS)
//   Y = P @ V^T (WMMA) -> f16 [4096,768] workspace
// LDS: 8K (Q) + 64K (S) + 64K (P) + ~3K  ~= 139 KB (CDNA5 WGP has 320 KB)
// ---------------------------------------------------------------------------
__global__ __launch_bounds__(256) void attn_kernel(
    const _Float16* __restrict__ Q,   // [B*H, T, 64]
    const _Float16* __restrict__ K,   // [B*H, T, 64]
    const _Float16* __restrict__ Vt,  // [B*H, 64, T]
    const _Float16* __restrict__ AF,  // [B*H, T, T]
    const float*    __restrict__ pad, // [B,T]
    const float*    __restrict__ adjw,// scalar
    _Float16*       __restrict__ Y)   // [4096, 768] f16
{
    __shared__ _Float16 Qs[64 * 64];
    __shared__ _Float16 Ss[64 * 512];
    __shared__ _Float16 Ps[64 * 512];
    __shared__ float    red[64 * 4];
    __shared__ float    rowmax[64];
    __shared__ float    rowsum[64];
    __shared__ float    keeps[512];

    const int bh = blockIdx.x;            // b*H + h
    const int b  = bh / Hn;
    const int h  = bh % Hn;
    const int q0 = blockIdx.y * 64;
    const int tid = threadIdx.x;
    const int wave = tid >> 5, lane = tid & 31;
    const int r16 = lane & 15, hlf = lane >> 4;

    const _Float16* Qb  = Q  + (size_t)bh * Tn * HDn;
    const _Float16* Kb  = K  + (size_t)bh * Tn * HDn;
    const _Float16* Vtb = Vt + (size_t)bh * HDn * Tn;

    // stage Q tile via CDNA5 async global->LDS copies + keep vector
    {
        const int row = tid >> 2, col = (tid & 3) * 16;
        const _Float16* src = Qb + (size_t)(q0 + row) * HDn + col;
        async_copy16(&Qs[row * 64 + col],     src);
        async_copy16(&Qs[row * 64 + col + 8], src + 8);
        for (int i = tid; i < Tn; i += 256) keeps[i] = 1.0f - pad[b * Tn + i];
        async_wait0();
    }
    __syncthreads();

    // ---- Phase 1: S = Q K^T.  wave -> Mtile (wave&3), N half (wave>>2)*256
    {
        const int mt = wave & 3;
        const int nbase = (wave >> 2) * 256;
        v16h afrag[2];
#pragma unroll
        for (int kk = 0; kk < 2; ++kk)
            afrag[kk] = load_frag_h(&Qs[(mt * 16 + r16) * 64 + kk * 32 + hlf * 8]);
#pragma unroll 4
        for (int nt = 0; nt < 16; ++nt) {
            const int n0 = nbase + nt * 16;
            // prefetch next key rows (gfx1250 global_prefetch_b8; speculative OK)
            __builtin_prefetch(Kb + (size_t)(n0 + 16 + r16) * HDn + hlf * 8, 0, 1);
            v8f acc = v8f{};
#pragma unroll
            for (int kk = 0; kk < 2; ++kk) {
                v16h bf = load_frag_h(Kb + (size_t)(n0 + r16) * HDn + kk * 32 + hlf * 8);
                acc = __builtin_amdgcn_wmma_f32_16x16x32_f16(
                          false, afrag[kk], false, bf, (short)0, acc, false, false);
            }
#pragma unroll
            for (int r = 0; r < 8; ++r)
                Ss[(mt * 16 + hlf * 8 + r) * 512 + n0 + r16] = (_Float16)acc[r];
        }
    }
    __syncthreads();

    // ---- Phase 2: masked softmax (f32) + adjacency bias, into Ps (f16)
    {
        const float scale = 0.125f;               // 1/sqrt(64)
        const float aw = adjw[0];
        const int row = tid >> 2, seg = tid & 3, c0 = seg * 128;
        const float keepq = keeps[q0 + row];

        float m = -1e30f;
        for (int j = 0; j < 128; ++j) {
            const int kc = c0 + j;
            const float s = (keepq * keeps[kc] == 0.f)
                              ? -1e9f : (float)Ss[row * 512 + kc] * scale;
            m = fmaxf(m, s);
        }
        red[row * 4 + seg] = m;
        __syncthreads();
        if (seg == 0)
            rowmax[row] = fmaxf(fmaxf(red[row * 4], red[row * 4 + 1]),
                                fmaxf(red[row * 4 + 2], red[row * 4 + 3]));
        __syncthreads();
        const float rm = rowmax[row];
        float sum = 0.f;
        for (int j = 0; j < 128; ++j) {
            const int kc = c0 + j;
            const float s = (keepq * keeps[kc] == 0.f)
                              ? -1e9f : (float)Ss[row * 512 + kc] * scale;
            const float e = __expf(s - rm);
            sum += e;
            Ps[row * 512 + kc] = (_Float16)e;
        }
        red[row * 4 + seg] = sum;
        __syncthreads();
        if (seg == 0)
            rowsum[row] = red[row * 4] + red[row * 4 + 1] +
                          red[row * 4 + 2] + red[row * 4 + 3];
        __syncthreads();
        const float inv = 1.0f / rowsum[row];
        const _Float16* afr = AF + ((size_t)bh * Tn + (q0 + row)) * Tn;
        for (int j = 0; j < 128; ++j) {
            const int kc = c0 + j;
            const float p = (float)Ps[row * 512 + kc] * inv + aw * (float)afr[kc];
            Ps[row * 512 + kc] = (_Float16)p;
        }
    }
    __syncthreads();

    // ---- Phase 3: Y = P @ V^T.  wave -> Mtile (wave&3), Ntiles {0,1}/{2,3}
    {
        const int mt = wave & 3;
        const int ntb = (wave >> 2) * 2;
        v8f acc[2] = {v8f{}, v8f{}};
#pragma unroll 4
        for (int kt = 0; kt < Tn / 32; ++kt) {
            const int k0 = kt * 32;
            // prefetch next V^T chunk
            __builtin_prefetch(Vtb + (size_t)(ntb * 16 + r16) * Tn + k0 + 32 + hlf * 8, 0, 1);
            v16h a = load_frag_h(&Ps[(mt * 16 + r16) * 512 + k0 + hlf * 8]);
#pragma unroll
            for (int i = 0; i < 2; ++i) {
                const int n0 = (ntb + i) * 16;
                v16h bf = load_frag_h(Vtb + (size_t)(n0 + r16) * Tn + k0 + hlf * 8);
                acc[i] = __builtin_amdgcn_wmma_f32_16x16x32_f16(
                             false, a, false, bf, (short)0, acc[i], false, false);
            }
        }
#pragma unroll
        for (int i = 0; i < 2; ++i) {
            const int n0 = (ntb + i) * 16;
#pragma unroll
            for (int r = 0; r < 8; ++r) {
                const size_t rowg = (size_t)b * Tn + q0 + mt * 16 + hlf * 8 + r;
                Y[rowg * Cn + h * HDn + n0 + r16] = (_Float16)acc[i][r];
            }
        }
    }
}

// ---------------------------------------------------------------------------
// Kernel 4: output projection.  out = Yw(f16) @ Wph(f16)^T + bp  (f32 out)
// ---------------------------------------------------------------------------
__global__ __launch_bounds__(256) void gemm_proj(
    const _Float16* __restrict__ Yw,    // [4096,768] f16
    const _Float16* __restrict__ Wp,    // [768,768] f16
    const float*    __restrict__ bp,    // [768]
    float*          __restrict__ Out)   // [4096,768] f32
{
    const int m0blk = blockIdx.x * 64;
    const int n0blk = blockIdx.y * 64;
    const int tid = threadIdx.x;
    const int wave = tid >> 5, lane = tid & 31;
    const int r16 = lane & 15, hlf = lane >> 4;
    const int mt  = wave & 3;
    const int ntb = (wave >> 2) * 2;

    const _Float16* arow  = Yw + (size_t)(m0blk + mt * 16 + r16) * Cn + hlf * 8;
    const _Float16* brow0 = Wp + (size_t)(n0blk + ntb * 16 + r16) * Cn + hlf * 8;
    const _Float16* brow1 = brow0 + (size_t)16 * Cn;

    v8f acc[2] = {v8f{}, v8f{}};
#pragma unroll 4
    for (int kt = 0; kt < Cn / 32; ++kt) {
        const int k0 = kt * 32;
        v16h a  = load_frag_h(arow + k0);
        v16h b0 = load_frag_h(brow0 + k0);
        v16h b1 = load_frag_h(brow1 + k0);
        acc[0] = __builtin_amdgcn_wmma_f32_16x16x32_f16(
                     false, a, false, b0, (short)0, acc[0], false, false);
        acc[1] = __builtin_amdgcn_wmma_f32_16x16x32_f16(
                     false, a, false, b1, (short)0, acc[1], false, false);
    }
#pragma unroll
    for (int i = 0; i < 2; ++i) {
        const int col = n0blk + (ntb + i) * 16 + r16;
        const float bv = bp[col];
#pragma unroll
        for (int r = 0; r < 8; ++r) {
            const size_t rowg = m0blk + mt * 16 + hlf * 8 + r;
            Out[rowg * Cn + col] = acc[i][r] + bv;
        }
    }
}

// ---------------------------------------------------------------------------
extern "C" void kernel_launch(void* const* d_in, const int* in_sizes, int n_in,
                              void* d_out, int out_size, void* d_ws, size_t ws_size,
                              hipStream_t stream)
{
    const float* x    = (const float*)d_in[0];
    const float* pad  = (const float*)d_in[1];
    const float* adj  = (const float*)d_in[2];
    const float* Wq   = (const float*)d_in[3];
    const float* bq   = (const float*)d_in[4];
    const float* Wk   = (const float*)d_in[5];
    const float* bk   = (const float*)d_in[6];
    const float* Wv   = (const float*)d_in[7];
    const float* bv   = (const float*)d_in[8];
    const float* Wp   = (const float*)d_in[9];
    const float* bp   = (const float*)d_in[10];
    const float* A1   = (const float*)d_in[11];
    const float* A2   = (const float*)d_in[12];
    const float* adjw = (const float*)d_in[13];
    float* out = (float*)d_out;

    // workspace layout (all f16)
    const size_t qkv_elems = (size_t)Bn * Hn * Tn * HDn;     // 3,145,728
    const size_t af_elems  = (size_t)Bn * Hn * Tn * Tn;      // 25,165,824
    const size_t y_elems   = (size_t)BT * Cn;                // 3,145,728
    const size_t x_elems   = (size_t)BT * Cn;                // 3,145,728
    const size_t w_elems   = (size_t)Cn * Cn;                //   589,824

    char* ws = (char*)d_ws;
    _Float16* Qw  = (_Float16*)ws;  ws += qkv_elems * sizeof(_Float16);
    _Float16* Kw  = (_Float16*)ws;  ws += qkv_elems * sizeof(_Float16);
    _Float16* Vtw = (_Float16*)ws;  ws += qkv_elems * sizeof(_Float16);
    _Float16* AFw = (_Float16*)ws;  ws += af_elems  * sizeof(_Float16);
    _Float16* Yw  = (_Float16*)ws;  ws += y_elems   * sizeof(_Float16);
    _Float16* Xh  = (_Float16*)ws;  ws += x_elems   * sizeof(_Float16);
    _Float16* Wqh = (_Float16*)ws;  ws += w_elems   * sizeof(_Float16);
    _Float16* Wkh = (_Float16*)ws;  ws += w_elems   * sizeof(_Float16);
    _Float16* Wvh = (_Float16*)ws;  ws += w_elems   * sizeof(_Float16);
    _Float16* Wph = (_Float16*)ws;  ws += w_elems   * sizeof(_Float16);

    // 0) one-time f32 -> f16 conversions (read once; GEMMs then stream f16)
    {
        const int nx4 = (int)(x_elems / 4), nw4 = (int)(w_elems / 4);
        cvt_f16_kernel<<<(nx4 + 255) / 256, 256, 0, stream>>>(x,  Xh,  nx4);
        cvt_f16_kernel<<<(nw4 + 255) / 256, 256, 0, stream>>>(Wq, Wqh, nw4);
        cvt_f16_kernel<<<(nw4 + 255) / 256, 256, 0, stream>>>(Wk, Wkh, nw4);
        cvt_f16_kernel<<<(nw4 + 255) / 256, 256, 0, stream>>>(Wv, Wvh, nw4);
        cvt_f16_kernel<<<(nw4 + 255) / 256, 256, 0, stream>>>(Wp, Wph, nw4);
    }

    // 1) adjacency bias (reads adj exactly once)
    adjbias_kernel<<<(Bn * Tn * Tn) / 256, 256, 0, stream>>>(adj, A1, A2, AFw);

    // 2) Q/K/V projections (WMMA)
    dim3 gGemm(BT / 64, Cn / 64);
    gemm_qkv<0><<<gGemm, 256, 0, stream>>>(Xh, Wqh, bq, Qw);
    gemm_qkv<0><<<gGemm, 256, 0, stream>>>(Xh, Wkh, bk, Kw);
    gemm_qkv<1><<<gGemm, 256, 0, stream>>>(Xh, Wvh, bv, Vtw);

    // 3) fused attention (WMMA + async-LDS staging + LDS softmax + bias)
    attn_kernel<<<dim3(Bn * Hn, Tn / 64), 256, 0, stream>>>(
        Qw, Kw, Vtw, AFw, pad, adjw, Yw);

    // 4) output projection (WMMA)
    gemm_proj<<<gGemm, 256, 0, stream>>>(Yw, Wph, bp, out);
}
